// MoeLayer_63084479643855
// MI455X (gfx1250) — matrix operand order repaired and checked by reading, not compiled, not run on hardware
//
#include <hip/hip_runtime.h>
#include <hip/hip_bf16.h>
#include <cstdint>

// ---------------- problem constants (match reference) ----------------
constexpr int KE = 8;      // experts
constexpr int KD = 512;    // d_model
constexpr int KF = 2048;   // d_ff
constexpr int BM = 32;     // tokens per block tile
constexpr int BF = 64;     // F chunk
constexpr int XS_STRIDE = KD + 16;   // bf16 elems, keep 16B-aligned rows, skew banks
constexpr int HS_STRIDE = BF + 8;    // bf16 elems

typedef __attribute__((ext_vector_type(16))) __bf16 v16bf;
typedef __attribute__((ext_vector_type(8)))  float  v8f;

union AFrag { v16bf v; uint4 q[2]; };

static __device__ __forceinline__ v8f wmma_bf16(const v16bf& a, const v16bf& b, const v8f& c) {
    return __builtin_amdgcn_wmma_f32_16x16x32_bf16(false, a, false, b, (short)0, c, false, false);
}

// fast silu: x * sigmoid(x) using v_rcp_f32 instead of IEEE divide
static __device__ __forceinline__ float fast_silu(float x) {
    return x * __builtin_amdgcn_rcpf(1.0f + __expf(-x));
}

// ---------------- x fp32 -> bf16 ----------------
__global__ void cvt_x_kernel(const float* __restrict__ src, __bf16* __restrict__ dst, size_t n) {
    size_t i = ((size_t)blockIdx.x * blockDim.x + threadIdx.x) * 4;
    if (i >= n) return;
    float4 v = *(const float4*)(src + i);
    union { __bf16 h[4]; uint2 u; } o;
    o.h[0] = (__bf16)v.x; o.h[1] = (__bf16)v.y; o.h[2] = (__bf16)v.z; o.h[3] = (__bf16)v.w;
    *(uint2*)(dst + i) = o.u;
}

// ---------------- transpose + convert: src [E][R][C] f32 -> dst [E][C][R] bf16 ----------------
__global__ void transpose_cvt_kernel(const float* __restrict__ src, __bf16* __restrict__ dst,
                                     int R, int C) {
    __shared__ float tile[32][33];
    const float* s = src + (size_t)blockIdx.z * R * C;
    __bf16*      d = dst + (size_t)blockIdx.z * R * C;
    const int c0 = blockIdx.x * 32, r0 = blockIdx.y * 32;
    const int tx = threadIdx.x, ty = threadIdx.y;   // 32 x 8
    #pragma unroll
    for (int i = 0; i < 32; i += 8)
        tile[ty + i][tx] = s[(size_t)(r0 + ty + i) * C + (c0 + tx)];
    __syncthreads();
    #pragma unroll
    for (int i = 0; i < 32; i += 8)
        d[(size_t)(c0 + ty + i) * R + (r0 + tx)] = (__bf16)tile[tx][ty + i];
}

// ---------------- gating: logits, top-2 softmax, build per-expert lists ----------------
__global__ void gate_kernel(const float* __restrict__ x, const float* __restrict__ gw,
                            const float* __restrict__ gb, int* __restrict__ counts,
                            int* __restrict__ idxl, float* __restrict__ wl, int N) {
    int t = blockIdx.x * blockDim.x + threadIdx.x;
    if (t >= N) return;
    float acc[KE];
    #pragma unroll
    for (int e = 0; e < KE; ++e) acc[e] = gb[e];
    const float* xr = x + (size_t)t * KD;
    for (int d = 0; d < KD; d += 4) {
        float4 xv = *(const float4*)(xr + d);
        const float* g = gw + (size_t)d * KE;
        #pragma unroll
        for (int e = 0; e < KE; ++e)
            acc[e] += xv.x * g[e] + xv.y * g[KE + e] + xv.z * g[2 * KE + e] + xv.w * g[3 * KE + e];
    }
    int e0 = 0; float v0 = acc[0];
    #pragma unroll
    for (int e = 1; e < KE; ++e) if (acc[e] > v0) { v0 = acc[e]; e0 = e; }
    int e1 = -1; float v1 = -3.0e38f;
    #pragma unroll
    for (int e = 0; e < KE; ++e) if (e != e0 && acc[e] > v1) { v1 = acc[e]; e1 = e; }
    float z  = __expf(v1 - v0);
    float p1 = z / (1.0f + z);
    float p0 = 1.0f - p1;
    int s0 = atomicAdd(&counts[e0], 1);
    idxl[(size_t)e0 * N + s0] = t;  wl[(size_t)e0 * N + s0] = p0;
    int s1 = atomicAdd(&counts[e1], 1);
    idxl[(size_t)e1 * N + s1] = t;  wl[(size_t)e1 * N + s1] = p1;
}

// ---------------- fused gathered expert FFN (SwiGLU) ----------------
// xb  : [N][D]  bf16 activations
// w1t : [E][F][D] bf16 (transposed)   w3t: same   w2t: [E][D][F] bf16 (transposed)
__global__ __launch_bounds__(256) void moe_expert_kernel(
    const __bf16* __restrict__ xb, const __bf16* __restrict__ w1t,
    const __bf16* __restrict__ w3t, const __bf16* __restrict__ w2t,
    const int* __restrict__ counts, const int* __restrict__ idxl,
    const float* __restrict__ wl, float* __restrict__ out, int N) {

    const int e   = blockIdx.y;
    const int cnt = counts[e];
    const int m0  = blockIdx.x * BM;
    if (m0 >= cnt) return;

    __shared__ __bf16 xs[BM * XS_STRIDE];
    __shared__ __bf16 hs[BM * HS_STRIDE];
    __shared__ int    sIdx[BM];
    __shared__ float  sW[BM];

    const int tid = threadIdx.x;
    if (tid < BM) {
        int r  = m0 + tid;
        int ok = r < cnt;
        sIdx[tid] = ok ? idxl[(size_t)e * N + r] : -1;
        sW[tid]   = ok ? wl[(size_t)e * N + r]   : 0.0f;
    }
    __syncthreads();

    // stage 32 gathered token rows: 32 x (512 bf16) = 2048 16B transfers, 8 per thread
    #pragma unroll
    for (int i = 0; i < 8; ++i) {
        int lin = tid + 256 * i;
        int r   = lin >> 6;
        int c   = (lin & 63) << 3;   // element offset (8 bf16 = 16B)
        uint4 val = make_uint4(0u, 0u, 0u, 0u);
        int t = sIdx[r];
        if (t >= 0) val = *(const uint4*)(xb + (size_t)t * KD + c);
        *(uint4*)(xs + r * XS_STRIDE + c) = val;
    }
    __syncthreads();

    const int wave   = tid >> 5;
    const int lane   = tid & 31;
    const int rowSel = lane & 15;
    const int hiSel  = lane >> 4;

    const __bf16* w1e = w1t + (size_t)e * KF * KD;
    const __bf16* w3e = w3t + (size_t)e * KF * KD;
    const __bf16* w2e = w2t + (size_t)e * KD * KF;

    // GEMM1: wave computes h-tile (mt1, nt1) for both w1 and w3 paths
    const int mt1 = wave >> 2;   // 0..1
    const int nt1 = wave & 3;    // 0..3
    // GEMM2: wave owns output columns [wave*64, wave*64+64)
    const int d0 = wave * 64;

    v8f acc[2][4];
    #pragma unroll
    for (int mt = 0; mt < 2; ++mt)
        #pragma unroll
        for (int nt = 0; nt < 4; ++nt)
            #pragma unroll
            for (int r = 0; r < 8; ++r) acc[mt][nt][r] = 0.0f;

    #pragma unroll 1
    for (int chunk = 0; chunk < KF / BF; ++chunk) {
        const int f0 = chunk * BF;

        // ---- GEMM1: c1 = x·w1 tile, c3 = x·w3 tile (16x16 each, K = 512)
        v8f c1, c3;
        #pragma unroll
        for (int r = 0; r < 8; ++r) { c1[r] = 0.0f; c3[r] = 0.0f; }
        {
            const __bf16* ap  = xs  + (mt1 * 16 + rowSel) * XS_STRIDE + hiSel * 8;
            const __bf16* b1p = w1e + (size_t)(f0 + nt1 * 16 + rowSel) * KD + hiSel * 8;
            const __bf16* b3p = w3e + (size_t)(f0 + nt1 * 16 + rowSel) * KD + hiSel * 8;
            #pragma unroll 4
            for (int k = 0; k < KD; k += 32) {
                AFrag a, b1, b3;
                a.q[0]  = *(const uint4*)(ap + k);
                a.q[1]  = *(const uint4*)(ap + k + 16);
                b1.q[0] = *(const uint4*)(b1p + k);
                b1.q[1] = *(const uint4*)(b1p + k + 16);
                b3.q[0] = *(const uint4*)(b3p + k);
                b3.q[1] = *(const uint4*)(b3p + k + 16);
                c1 = wmma_bf16(a.v, b1.v, c1);
                c3 = wmma_bf16(a.v, b3.v, c3);
            }
        }
        // ---- SwiGLU elementwise, h -> LDS (bf16), fast sigmoid via v_rcp_f32
        #pragma unroll
        for (int r = 0; r < 8; ++r) {
            float s = fast_silu(c1[r]) * c3[r];
            int row = mt1 * 16 + r + hiSel * 8;
            int col = nt1 * 16 + rowSel;
            hs[row * HS_STRIDE + col] = (__bf16)s;
        }
        __syncthreads();

        // ---- GEMM2 partial: acc += h[32xBF] · w2[BF x 64-slice]
        #pragma unroll
        for (int k = 0; k < BF; k += 32) {
            AFrag a0, a1;
            const __bf16* hap = hs + rowSel * HS_STRIDE + k + hiSel * 8;
            a0.q[0] = *(const uint4*)(hap);
            a0.q[1] = *(const uint4*)(hap + 16);
            a1.q[0] = *(const uint4*)(hap + 16 * HS_STRIDE);
            a1.q[1] = *(const uint4*)(hap + 16 * HS_STRIDE + 16);
            #pragma unroll
            for (int nt = 0; nt < 4; ++nt) {
                const __bf16* bp = w2e + (size_t)(d0 + nt * 16 + rowSel) * KF + f0 + k + hiSel * 8;
                AFrag b;
                b.q[0] = *(const uint4*)(bp);
                b.q[1] = *(const uint4*)(bp + 16);
                acc[0][nt] = wmma_bf16(a0.v, b.v, acc[0][nt]);
                acc[1][nt] = wmma_bf16(a1.v, b.v, acc[1][nt]);
            }
        }
        __syncthreads();
    }

    // ---- weighted scatter-add to out (tokens may belong to 2 experts)
    #pragma unroll
    for (int mt = 0; mt < 2; ++mt)
        #pragma unroll
        for (int nt = 0; nt < 4; ++nt)
            #pragma unroll
            for (int r = 0; r < 8; ++r) {
                int row = mt * 16 + r + hiSel * 8;
                int t = sIdx[row];
                if (t >= 0) {
                    int col = d0 + nt * 16 + rowSel;
                    atomicAdd(out + (size_t)t * KD + col, sW[row] * acc[mt][nt][r]);
                }
            }
}

// ---------------- launcher ----------------
static inline size_t align256(size_t x) { return (x + 255) & ~(size_t)255; }

extern "C" void kernel_launch(void* const* d_in, const int* in_sizes, int n_in,
                              void* d_out, int out_size, void* d_ws, size_t ws_size,
                              hipStream_t stream) {
    const float* x   = (const float*)d_in[0];   // [N, 512]
    const float* gw  = (const float*)d_in[1];   // [512, 8]
    const float* gb  = (const float*)d_in[2];   // [8]
    const float* w1  = (const float*)d_in[3];   // [8, 512, 2048]
    const float* w3  = (const float*)d_in[4];   // [8, 512, 2048]
    const float* w2  = (const float*)d_in[5];   // [8, 2048, 512]
    float* out = (float*)d_out;

    const int N = in_sizes[0] / KD;             // 34816 tokens

    // workspace carve-out
    char* p = (char*)d_ws;
    __bf16* xb  = (__bf16*)p;  p += align256((size_t)N * KD * sizeof(__bf16));
    __bf16* w1t = (__bf16*)p;  p += align256((size_t)KE * KD * KF * sizeof(__bf16));
    __bf16* w3t = (__bf16*)p;  p += align256((size_t)KE * KD * KF * sizeof(__bf16));
    __bf16* w2t = (__bf16*)p;  p += align256((size_t)KE * KD * KF * sizeof(__bf16));
    int*    counts = (int*)p;  p += 256;
    int*    idxl   = (int*)p;  p += align256((size_t)KE * N * sizeof(int));
    float*  wl     = (float*)p;

    hipMemsetAsync(d_out, 0, (size_t)N * KD * sizeof(float), stream);
    hipMemsetAsync(counts, 0, KE * sizeof(int), stream);

    // convert activations to bf16
    {
        size_t n = (size_t)N * KD;
        int blocks = (int)((n / 4 + 255) / 256);
        cvt_x_kernel<<<blocks, 256, 0, stream>>>(x, xb, n);
    }
    // convert + transpose weights: w1/w3 [E][D][F] -> [E][F][D]; w2 [E][F][D] -> [E][D][F]
    transpose_cvt_kernel<<<dim3(KF / 32, KD / 32, KE), dim3(32, 8), 0, stream>>>(w1, w1t, KD, KF);
    transpose_cvt_kernel<<<dim3(KF / 32, KD / 32, KE), dim3(32, 8), 0, stream>>>(w3, w3t, KD, KF);
    transpose_cvt_kernel<<<dim3(KD / 32, KF / 32, KE), dim3(32, 8), 0, stream>>>(w2, w2t, KF, KD);

    // gating + routing
    gate_kernel<<<(N + 255) / 256, 256, 0, stream>>>(x, gw, gb, counts, idxl, wl, N);

    // fused gathered expert FFN
    dim3 grid((N + BM - 1) / BM, KE);
    moe_expert_kernel<<<grid, 256, 0, stream>>>(xb, w1t, w3t, w2t, counts, idxl, wl, out, N);
}
